// GumbelSelector_19550691131588
// MI455X (gfx1250) — matrix operand order repaired and checked by reading, not compiled, not run on hardware
//
#include <hip/hip_runtime.h>

typedef __attribute__((ext_vector_type(16))) _Float16 v16h;
typedef __attribute__((ext_vector_type(8)))  _Float16 v8h;
typedef __attribute__((ext_vector_type(8)))  float    v8f;

#define D_IN   256
#define D_HID  128
#define N_SLOT 4096
#define B_ROWS 64
#define M_TOT  (B_ROWS * N_SLOT)   // 262144 rows total
#define NT_CNT 8                   // 128 / 16 N-tiles
#define KT_CNT 8                   // 256 / 32 K-steps
#define LDS_K  264                 // padded K stride in halves (bank-conflict-free)

// ---------------- init: zero per-batch-row reduction state ----------------
__global__ void __launch_bounds__(64)
gumbel_init(unsigned long long* wsKey, unsigned int* wsCnt) {
  int t = threadIdx.x;
  if (t < B_ROWS) { wsKey[t] = 0ull; wsCnt[t] = 0u; }
}

// ---------------- main: fused MLP + decision + softmax --------------------
// waves_per_eu >= 3 (~336 VGPR cap, 3 waves/SIMD). The kt loop is kept ROLLED
// so per-iteration live values stay bounded (~220 VGPRs) and the allocator
// never spills (round-2 build spilled when kt was fully unrolled).
__global__ void __launch_bounds__(128, 3)
gumbel_mlp(const float* __restrict__ s, const float* __restrict__ W1,
           const float* __restrict__ b1, const float* __restrict__ W2,
           const float* __restrict__ b2, const float* __restrict__ rnoise,
           float* __restrict__ dec_out, float* __restrict__ prob_out,
           unsigned long long* __restrict__ wsKey, unsigned int* __restrict__ wsCnt)
{
  // W1 staged transposed [n][k] as f16; K-stride padded: 264 halves = 132 dwords,
  // 132 mod 64 = 4 -> 16 lanes of a B-fragment load land on distinct bank groups.
  __shared__ _Float16 sW1[D_HID][LDS_K];

  const int tid  = threadIdx.x;
  const int wave = tid >> 5;
  const int lane = tid & 31;
  const int lh   = lane & 15;   // lane within half-wave
  const int hi   = lane >> 4;   // which half-wave

  // Stage W1 (fp32 row-major [k][n]) -> LDS f16 [n][k]; coalesced global reads.
  for (int i = tid; i < D_IN * D_HID; i += 128) {
    int k = i >> 7;       // i / 128
    int n = i & 127;      // i % 128
    sW1[n][k] = (_Float16)W1[i];
  }
  __syncthreads();

  // Per-lane constants for bias + tiny 128->2 GEMM (column n = nt*16 + lh).
  float b1r[NT_CNT], w20[NT_CNT], w21[NT_CNT];
#pragma unroll
  for (int nt = 0; nt < NT_CNT; ++nt) {
    int n = nt * 16 + lh;
    b1r[nt] = b1[n];
    w20[nt] = W2[n * 2 + 0];
    w21[nt] = W2[n * 2 + 1];
  }
  const float b20 = b2[0], b21 = b2[1];

  const int rowBase0 = blockIdx.x * 256 + wave * 64;  // 64 consecutive rows per wave
  const int brow     = rowBase0 >> 12;                // batch row (4096 slots each)

  // ---- per-batch-row rnoise argmax: packed (bits(rnoise)<<32)|slot key ----
  {
    int g0 = rowBase0 + lane, g1 = rowBase0 + 32 + lane;
    unsigned long long k0 =
        ((unsigned long long)__float_as_uint(rnoise[g0]) << 32) | (unsigned)(g0 & (N_SLOT - 1));
    unsigned long long k1 =
        ((unsigned long long)__float_as_uint(rnoise[g1]) << 32) | (unsigned)(g1 & (N_SLOT - 1));
    unsigned long long km = (k0 > k1) ? k0 : k1;
#pragma unroll
    for (int off = 16; off >= 1; off >>= 1) {
      unsigned long long o = __shfl_xor(km, off, 32);
      km = (o > km) ? o : km;
    }
    if (lane == 0) atomicMax(&wsKey[brow], km);
  }

  unsigned int myCount = 0;

  // Two 32-row macro-tiles per wave: each B fragment feeds TWO wmmas (rows
  // 0..15 and 16..31), halving LDS traffic per output row and doubling ILP.
#pragma unroll 1
  for (int t = 0; t < 2; ++t) {
    const int rowBase = rowBase0 + t * 32;
    const int m0 = rowBase + lh;            // A-fragment-0 row for this lane
    const float* __restrict__ srow0 = s + (size_t)m0 * D_IN;
    const float* __restrict__ srow1 = srow0 + 16 * D_IN;   // A-fragment-1 row

    if (rowBase + 32 < M_TOT)
      __builtin_prefetch(srow0 + 32 * D_IN, 0, 0);   // global_prefetch_b8: next tile

    v8f acc0[NT_CNT], acc1[NT_CNT];
#pragma unroll
    for (int nt = 0; nt < NT_CNT; ++nt) { acc0[nt] = (v8f)(0.0f); acc1[nt] = (v8f)(0.0f); }

    // ROLLED k-loop: bounded register pressure, scheduler pipelines ~1 step.
#pragma unroll 1
    for (int kt = 0; kt < KT_CNT; ++kt) {
      // A fragments (16x32 f16): lane (hi,lh) holds K = kt*32 + 8*hi + {0..7, 16..23}.
      const float* ap0 = srow0 + kt * 32 + 8 * hi;
      const float* ap1 = srow1 + kt * 32 + 8 * hi;
      float4 a0 = *(const float4*)(ap0 + 0);
      float4 a1 = *(const float4*)(ap0 + 4);
      float4 a2 = *(const float4*)(ap0 + 16);
      float4 a3 = *(const float4*)(ap0 + 20);
      float4 c0 = *(const float4*)(ap1 + 0);
      float4 c1 = *(const float4*)(ap1 + 4);
      float4 c2 = *(const float4*)(ap1 + 16);
      float4 c3 = *(const float4*)(ap1 + 20);
      v16h af0, af1;
      af0[0]=(_Float16)a0.x; af0[1]=(_Float16)a0.y; af0[2]=(_Float16)a0.z; af0[3]=(_Float16)a0.w;
      af0[4]=(_Float16)a1.x; af0[5]=(_Float16)a1.y; af0[6]=(_Float16)a1.z; af0[7]=(_Float16)a1.w;
      af0[8]=(_Float16)a2.x; af0[9]=(_Float16)a2.y; af0[10]=(_Float16)a2.z; af0[11]=(_Float16)a2.w;
      af0[12]=(_Float16)a3.x; af0[13]=(_Float16)a3.y; af0[14]=(_Float16)a3.z; af0[15]=(_Float16)a3.w;
      af1[0]=(_Float16)c0.x; af1[1]=(_Float16)c0.y; af1[2]=(_Float16)c0.z; af1[3]=(_Float16)c0.w;
      af1[4]=(_Float16)c1.x; af1[5]=(_Float16)c1.y; af1[6]=(_Float16)c1.z; af1[7]=(_Float16)c1.w;
      af1[8]=(_Float16)c2.x; af1[9]=(_Float16)c2.y; af1[10]=(_Float16)c2.z; af1[11]=(_Float16)c2.w;
      af1[12]=(_Float16)c3.x; af1[13]=(_Float16)c3.y; af1[14]=(_Float16)c3.z; af1[15]=(_Float16)c3.w;

#pragma unroll
      for (int nt = 0; nt < NT_CNT; ++nt) {
        // B fragment (32x16 f16): lane holds col = lh, K = kt*32 + 8*hi + {0..7, 16..23}.
        const v8h* bp = (const v8h*)&sW1[nt * 16 + lh][kt * 32 + 8 * hi];
        v8h blo = bp[0], bhi2 = bp[2];
        v16h bf;
#pragma unroll
        for (int e = 0; e < 8; ++e) { bf[e] = blo[e]; bf[8 + e] = bhi2[e]; }
        acc0[nt] = __builtin_amdgcn_wmma_f32_16x16x32_f16(
            false, af0, false, bf, (short)0, acc0[nt], false, false);
        acc1[nt] = __builtin_amdgcn_wmma_f32_16x16x32_f16(
            false, af1, false, bf, (short)0, acc1[nt], false, false);
      }
    }

    // Epilogue: bias + ReLU + 128->2 GEMM + argmax/softmax.
    // C/D layout: lane = col n, accumulator VGPR r = row (r + 8*hi).
#pragma unroll
    for (int half = 0; half < 2; ++half) {
      const v8f* acc = half ? acc1 : acc0;
      const int rbase = rowBase + half * 16;
#pragma unroll
      for (int r = 0; r < 8; ++r) {
        float p0 = 0.f, p1 = 0.f;
#pragma unroll
        for (int nt = 0; nt < NT_CNT; ++nt) {
          float h = acc[nt][r] + b1r[nt];
          h = (h > 0.f) ? h : 0.f;
          p0 += h * w20[nt];
          p1 += h * w21[nt];
        }
#pragma unroll
        for (int off = 8; off >= 1; off >>= 1) {   // reduce over 16 cols per half-wave
          p0 += __shfl_xor(p0, off, 16);
          p1 += __shfl_xor(p1, off, 16);
        }
        if (lh == 0) {
          int mr = rbase + r + hi * 8;
          float lo0 = p0 + b20, lo1 = p1 + b21;
          float dec = (lo1 > lo0) ? 1.0f : 0.0f;   // argmax == 1 (strict, ties -> 0)
          dec_out[mr]  = dec;
          prob_out[mr] = 1.0f / (1.0f + __expf(lo0 - lo1));  // softmax[...,1]
          myCount += (unsigned)dec;
        }
      }
    }
  }

  if (lh == 0 && myCount) atomicAdd(&wsCnt[brow], myCount);
}

// ---------------- rescue: force-on max-rnoise slot for empty rows ---------
__global__ void __launch_bounds__(64)
gumbel_rescue(const unsigned long long* __restrict__ wsKey,
              const unsigned int* __restrict__ wsCnt,
              float* __restrict__ dec_out) {
  int b = threadIdx.x;
  if (b < B_ROWS && wsCnt[b] == 0u) {
    unsigned n = (unsigned)(wsKey[b] & 0xffffffffu);
    dec_out[(size_t)b * N_SLOT + n] = 1.0f;
  }
}

extern "C" void kernel_launch(void* const* d_in, const int* in_sizes, int n_in,
                              void* d_out, int out_size, void* d_ws, size_t ws_size,
                              hipStream_t stream) {
  const float* s      = (const float*)d_in[0];
  const float* W1     = (const float*)d_in[1];
  const float* b1     = (const float*)d_in[2];
  const float* W2     = (const float*)d_in[3];
  const float* b2     = (const float*)d_in[4];
  const float* rnoise = (const float*)d_in[5];

  float* dec_out  = (float*)d_out;            // [B, N] flattened
  float* prob_out = dec_out + M_TOT;          // [B, N] flattened

  unsigned long long* wsKey = (unsigned long long*)d_ws;          // 64 * 8 B
  unsigned int*       wsCnt = (unsigned int*)((char*)d_ws + B_ROWS * sizeof(unsigned long long));

  gumbel_init<<<1, 64, 0, stream>>>(wsKey, wsCnt);
  gumbel_mlp<<<M_TOT / 256, 128, 0, stream>>>(s, W1, b1, W2, b2, rnoise,
                                              dec_out, prob_out, wsKey, wsCnt);
  gumbel_rescue<<<1, 64, 0, stream>>>(wsKey, wsCnt, dec_out);
}
